// MyLSTM_25022479467077
// MI455X (gfx1250) — matrix operand order repaired and checked by reading, not compiled, not run on hardware
//
#include <hip/hip_runtime.h>
#include <hip/hip_bf16.h>

// ---------------------------------------------------------------------------
// Bidirectional 2-layer LSTM for MI455X (gfx1250), wave32 + WMMA bf16.
//   T=1024, B=64, I=512, H=512.
// Strategy:
//   * Convert x and all weights to bf16 once (f32 accumulate in WMMA).
//   * 4 sequential direction passes (l0-fwd, l0-bwd, l1-fwd, l1-bwd).
//   * Per time step: one kernel, grid = (B/32, H/16) = 64 WGs, block = 128
//     (4 waves). Wave g computes TWO 16x16 tiles (rows b0..b0+31) of gate g
//     via v_wmma_f32_16x16x32_bf16 over K = I + H, sharing each B (weight)
//     operand between the two WMMAs -> halves weight L2 traffic on the
//     serial critical path. Gates combine through LDS, then the pointwise
//     sigmoid/tanh update produces c,h; h is double-buffered bf16.
// ---------------------------------------------------------------------------

typedef __attribute__((ext_vector_type(8)))  __bf16 v8bf;
typedef __attribute__((ext_vector_type(16))) __bf16 v16bf;
typedef __attribute__((ext_vector_type(8)))  float  v8f;
typedef __attribute__((ext_vector_type(4)))  float  v4f;
typedef __attribute__((ext_vector_type(4)))  __bf16 v4bf;

constexpr int kT = 1024;
constexpr int kB = 64;
constexpr int kI = 512;
constexpr int kH = 512;

static __device__ __forceinline__ v16bf cat8(v8bf lo, v8bf hi) {
  return __builtin_shufflevector(lo, hi, 0, 1, 2, 3, 4, 5, 6, 7,
                                 8, 9, 10, 11, 12, 13, 14, 15);
}

// ----------------------------- small utility kernels -----------------------

// n must be a multiple of 4 (all our arrays are).
__global__ void cvt_f32_to_bf16_v4(__bf16* __restrict__ dst,
                                   const float* __restrict__ src, long n4) {
  long i = (long)blockIdx.x * blockDim.x + threadIdx.x;
  long stride = (long)gridDim.x * blockDim.x;
  for (; i < n4; i += stride) {
    v4f v = *(const v4f*)(src + i * 4);
    v4bf o;
    o[0] = (__bf16)v[0];
    o[1] = (__bf16)v[1];
    o[2] = (__bf16)v[2];
    o[3] = (__bf16)v[3];
    *(v4bf*)(dst + i * 4) = o;
  }
}

__global__ void init_state(__bf16* __restrict__ hbf,
                           float* __restrict__ cst,
                           const float* __restrict__ h0,
                           const float* __restrict__ c0) {
  int i = blockIdx.x * blockDim.x + threadIdx.x;
  if (i < kB * kH) {
    hbf[i] = (__bf16)h0[i];
    cst[i] = c0[i];
  }
}

// Write final h/c of one direction pass into d_out (hn/cn regions).
__global__ void write_states(float* __restrict__ out,
                             const float* __restrict__ hstate,
                             const float* __restrict__ cstate,
                             int layer, int dir) {
  int i = blockIdx.x * blockDim.x + threadIdx.x;
  if (i >= kB * kH) return;
  int b = i / kH, j = i % kH;
  const long inputs_sz = (long)kT * kB * 2 * kH;
  const long hn_sz = (long)2 * kB * 2 * kH;
  long off = (long)layer * kB * 2 * kH + (long)b * 2 * kH + (long)dir * kH + j;
  out[inputs_sz + off] = hstate[i];
  out[inputs_sz + hn_sz + off] = cstate[i];
}

// ------------------------------- LSTM step ---------------------------------
// One time step of one direction.
//   xin : bf16 [kB][ldx]   (x slice for this t; layer0: x, layer1: out0)
//   Wih : bf16 [4H][ldx]   row-major -> directly WMMA-B (B = W^T) layout
//   Whh : bf16 [4H][kH]
//   bias: f32  [4H]
//   hin/hout: bf16 [kB][kH] double buffer
//   hstate/cstate: f32 [kB][kH] (persistent across steps; c updated in place)
//   out_bf / out_f: one non-null; [kB][ld_out] slice for this t (col offset
//                   for direction baked into the pointer)
__global__ __launch_bounds__(128) void lstm_step(
    const __bf16* __restrict__ xin, int ldx,
    const __bf16* __restrict__ Wih,
    const __bf16* __restrict__ Whh,
    const float* __restrict__ bias,
    const __bf16* __restrict__ hin,
    __bf16* __restrict__ hout,
    float* __restrict__ hstate,
    float* __restrict__ cstate,
    __bf16* __restrict__ out_bf,
    float* __restrict__ out_f,
    int ld_out) {
  const int tid = threadIdx.x;
  const int lane = tid & 31;
  const int wave = tid >> 5;        // gate: 0=i 1=f 2=g 3=o
  const int lc = lane & 15;         // A row / B col / C col
  const int hi = lane >> 4;         // half-wave selector

  const int b0 = blockIdx.x * 32;          // two 16-row tiles: b0, b0+16
  const int htile = blockIdx.y;            // 0..kH/16-1
  const int n0 = wave * kH + htile * 16;   // column base within 4H

  // C init = bias broadcast (C layout: col = lane&15, any row/vgpr)
  v8f acc0, acc1;
  {
    float bv = bias[n0 + lc];
#pragma unroll
    for (int r = 0; r < 8; ++r) { acc0[r] = bv; acc1[r] = bv; }
  }

  // ---- x contribution: K = ldx ----
  {
    const __bf16* Arow0 = xin + (long)(b0 + lc) * ldx + hi * 8;
    const __bf16* Arow1 = Arow0 + (long)16 * ldx;
    const __bf16* Brow = Wih + (long)(n0 + lc) * ldx + hi * 8;
#pragma unroll 4
    for (int kk = 0; kk < ldx; kk += 32) {
      v8bf bq0 = *(const v8bf*)(Brow + kk);
      v8bf bq1 = *(const v8bf*)(Brow + kk + 16);
      v16bf Bm = cat8(bq0, bq1);
      v8bf a00 = *(const v8bf*)(Arow0 + kk);
      v8bf a01 = *(const v8bf*)(Arow0 + kk + 16);
      v8bf a10 = *(const v8bf*)(Arow1 + kk);
      v8bf a11 = *(const v8bf*)(Arow1 + kk + 16);
      acc0 = __builtin_amdgcn_wmma_f32_16x16x32_bf16(
          false, cat8(a00, a01), false, Bm, (short)0, acc0, false, false);
      acc1 = __builtin_amdgcn_wmma_f32_16x16x32_bf16(
          false, cat8(a10, a11), false, Bm, (short)0, acc1, false, false);
    }
  }

  // ---- h contribution: K = kH ----
  {
    const __bf16* Arow0 = hin + (long)(b0 + lc) * kH + hi * 8;
    const __bf16* Arow1 = Arow0 + (long)16 * kH;
    const __bf16* Brow = Whh + (long)(n0 + lc) * kH + hi * 8;
#pragma unroll 4
    for (int kk = 0; kk < kH; kk += 32) {
      v8bf bq0 = *(const v8bf*)(Brow + kk);
      v8bf bq1 = *(const v8bf*)(Brow + kk + 16);
      v16bf Bm = cat8(bq0, bq1);
      v8bf a00 = *(const v8bf*)(Arow0 + kk);
      v8bf a01 = *(const v8bf*)(Arow0 + kk + 16);
      v8bf a10 = *(const v8bf*)(Arow1 + kk);
      v8bf a11 = *(const v8bf*)(Arow1 + kk + 16);
      acc0 = __builtin_amdgcn_wmma_f32_16x16x32_bf16(
          false, cat8(a00, a01), false, Bm, (short)0, acc0, false, false);
      acc1 = __builtin_amdgcn_wmma_f32_16x16x32_bf16(
          false, cat8(a10, a11), false, Bm, (short)0, acc1, false, false);
    }
  }

  // ---- gates meet in LDS (C layout: row = r + 8*hi, col = lc) ----
  __shared__ float gt[4][32][16];
#pragma unroll
  for (int r = 0; r < 8; ++r) {
    gt[wave][r + hi * 8][lc] = acc0[r];
    gt[wave][r + hi * 8 + 16][lc] = acc1[r];
  }
  __syncthreads();

  // ---- pointwise LSTM cell update: 512 elements over 128 threads ----
#pragma unroll
  for (int e = tid; e < 512; e += 128) {
    const int rr = e >> 4, cc = e & 15;
    const int b = b0 + rr;
    const int j = htile * 16 + cc;
    const long idx = (long)b * kH + j;
    const float si = 1.0f / (1.0f + __expf(-gt[0][rr][cc]));
    const float sf = 1.0f / (1.0f + __expf(-gt[1][rr][cc]));
    const float tg = tanhf(gt[2][rr][cc]);
    const float so = 1.0f / (1.0f + __expf(-gt[3][rr][cc]));
    const float c2 = sf * cstate[idx] + si * tg;
    const float h2 = so * tanhf(c2);
    cstate[idx] = c2;
    hstate[idx] = h2;
    hout[idx] = (__bf16)h2;
    const long oidx = (long)b * ld_out + j;
    if (out_bf) out_bf[oidx] = (__bf16)h2;
    if (out_f) out_f[oidx] = h2;
  }
}

// ------------------------------- host side ---------------------------------

extern "C" void kernel_launch(void* const* d_in, const int* in_sizes, int n_in,
                              void* d_out, int out_size, void* d_ws,
                              size_t ws_size, hipStream_t stream) {
  (void)in_sizes; (void)n_in; (void)out_size; (void)ws_size;

  const float* x = (const float*)d_in[0];
  const float* h0 = (const float*)d_in[1];
  const float* c0 = (const float*)d_in[2];
  // pass order: l0-fwd, l0-bwd, l1-fwd, l1-bwd
  const float* Wih_f32[4] = {(const float*)d_in[3], (const float*)d_in[6],
                             (const float*)d_in[9], (const float*)d_in[12]};
  const float* Whh_f32[4] = {(const float*)d_in[4], (const float*)d_in[7],
                             (const float*)d_in[10], (const float*)d_in[13]};
  const float* bias[4] = {(const float*)d_in[5], (const float*)d_in[8],
                          (const float*)d_in[11], (const float*)d_in[14]};

  // ---- carve workspace (256B aligned) ----
  char* p = (char*)d_ws;
  auto carve = [&](size_t bytes) -> char* {
    char* r = p;
    p += (bytes + 255) & ~(size_t)255;
    return r;
  };
  __bf16* xbf  = (__bf16*)carve((size_t)kT * kB * kI * 2);       // 64 MB
  __bf16* out0 = (__bf16*)carve((size_t)kT * kB * 2 * kH * 2);   // 128 MB
  __bf16* WihB[4];
  __bf16* WhhB[4];
  size_t wih_elems[4] = {(size_t)4 * kH * kI, (size_t)4 * kH * kI,
                         (size_t)4 * kH * 2 * kH, (size_t)4 * kH * 2 * kH};
  for (int d = 0; d < 4; ++d) WihB[d] = (__bf16*)carve(wih_elems[d] * 2);
  for (int d = 0; d < 4; ++d) WhhB[d] = (__bf16*)carve((size_t)4 * kH * kH * 2);
  __bf16* hbufA = (__bf16*)carve((size_t)kB * kH * 2);
  __bf16* hbufB = (__bf16*)carve((size_t)kB * kH * 2);
  float* hstate = (float*)carve((size_t)kB * kH * 4);
  float* cstate = (float*)carve((size_t)kB * kH * 4);

  // ---- one-time conversions to bf16 (vectorized x4) ----
  {
    const int th = 256, bl = 4096;
    cvt_f32_to_bf16_v4<<<bl, th, 0, stream>>>(xbf, x,
                                              (long)kT * kB * kI / 4);
    for (int d = 0; d < 4; ++d) {
      cvt_f32_to_bf16_v4<<<bl, th, 0, stream>>>(WihB[d], Wih_f32[d],
                                                (long)wih_elems[d] / 4);
      cvt_f32_to_bf16_v4<<<bl, th, 0, stream>>>(WhhB[d], Whh_f32[d],
                                                (long)4 * kH * kH / 4);
    }
  }

  float* out_final = (float*)d_out;  // [T][B][2H] then hn then cn

  // ---- 4 direction passes ----
  for (int pass = 0; pass < 4; ++pass) {
    const int layer = pass >> 1;
    const int rev = pass & 1;
    const int ldx = layer ? 2 * kH : kI;
    const __bf16* xin_base = layer ? out0 : xbf;

    init_state<<<(kB * kH + 255) / 256, 256, 0, stream>>>(hbufA, cstate, h0,
                                                          c0);

    dim3 grid(kB / 32, kH / 16);  // (2, 32) = 64 WGs
    for (int tau = 0; tau < kT; ++tau) {
      const int t = rev ? (kT - 1 - tau) : tau;
      const __bf16* hin = (tau & 1) ? hbufB : hbufA;
      __bf16* hout = (tau & 1) ? hbufA : hbufB;
      __bf16* obf = nullptr;
      float* of = nullptr;
      const long orow = (long)t * kB * 2 * kH + (rev ? kH : 0);
      if (layer == 0)
        obf = out0 + orow;
      else
        of = out_final + orow;
      lstm_step<<<grid, 128, 0, stream>>>(
          xin_base + (long)t * kB * ldx, ldx, WihB[pass], WhhB[pass],
          bias[pass], hin, hout, hstate, cstate, obf, of, 2 * kH);
    }

    write_states<<<(kB * kH + 255) / 256, 256, 0, stream>>>(out_final, hstate,
                                                            cstate, layer, rev);
  }
}
